// GraphConvClf2_67327907332511
// MI455X (gfx1250) — compile-verified, matrix operand order checked by reading, not run on hardware
//
#include <hip/hip_runtime.h>
#include <hip/hip_bf16.h>
#include <cstdint>

#define N_VERTS   200000
#define N_EDGES   600000
#define NUM_MESHES 32
#define DMID      128
#define NN1       256
#define BN_EPS    1e-5f

typedef __attribute__((ext_vector_type(16))) __bf16 v16bf;
typedef __attribute__((ext_vector_type(8)))  __bf16 v8bf;
typedef __attribute__((ext_vector_type(8)))  float  v8f;

static __device__ inline v16bf cat16(v8bf lo, v8bf hi) {
    return __builtin_shufflevector(lo, hi, 0,1,2,3,4,5,6,7,8,9,10,11,12,13,14,15);
}

// Per ISA 7.12.2 (16-bit A 16x32): for lane half h, VGPRs 0-3 hold K = K0 + h*8 + [0..7],
// VGPRs 4-7 hold K = K0 + 16 + h*8 + [0..7]; two contiguous 16-byte bf16 runs.
static __device__ inline v16bf load_frag_bf16(const __bf16* __restrict__ row, int K0, int half) {
    v8bf lo = *(const v8bf*)(row + K0 + half * 8);
    v8bf hi = *(const v8bf*)(row + K0 + 16 + half * 8);
    return cat16(lo, hi);
}

// Dual GEMM: H0 = X@W0^T + b0, H1 = X@W1^T + b1.
// X: N x 128 (bf16), W: 128 x 128 (bf16, pre-converted). Block = 256 threads = 8 waves.
// Both weight matrices are staged once per block into LDS with async global->LDS copies
// (global_load_async_to_lds_b128 / s_wait_asynccnt), then WMMA B-fragments come from LDS.
__global__ __launch_bounds__(256) void gemm_dual_wmma(
    const __bf16* __restrict__ Xbf,
    const __bf16* __restrict__ W0bf, const float* __restrict__ B0,
    const __bf16* __restrict__ W1bf, const float* __restrict__ B1,
    float* __restrict__ H0, float* __restrict__ H1)
{
    __shared__ __align__(16) __bf16 lds_w[2][DMID * DMID];   // 2 x 32 KB

    // ---- async stage W0,W1 into LDS: 2048 16-byte chunks per matrix, 256 lanes x 8 iters ----
    {
        const uint32_t t = threadIdx.x;
        uint32_t ldsbase = (uint32_t)(uintptr_t)(&lds_w[0][0]);
        uint64_t g0 = (uint64_t)(uintptr_t)W0bf;
        uint64_t g1 = (uint64_t)(uintptr_t)W1bf;
        #pragma unroll
        for (int i = 0; i < 8; ++i) {
            uint32_t byteoff = (uint32_t)(i * 256 + t) * 16u;
            uint32_t l0 = ldsbase + byteoff;
            uint32_t l1 = ldsbase + (uint32_t)(DMID * DMID * 2) + byteoff;
            uint64_t a0 = g0 + byteoff;
            uint64_t a1 = g1 + byteoff;
            asm volatile("global_load_async_to_lds_b128 %0, %1, off"
                         :: "v"(l0), "v"(a0) : "memory");
            asm volatile("global_load_async_to_lds_b128 %0, %1, off"
                         :: "v"(l1), "v"(a1) : "memory");
        }
        asm volatile("s_wait_asynccnt 0" ::: "memory");
    }
    __syncthreads();

    const int lane = threadIdx.x & 31;
    const int wave = threadIdx.x >> 5;
    const int mat  = wave >> 2;
    const int rowBase = blockIdx.x * 64 + (wave & 3) * 16;
    const int half = lane >> 4;
    const int l16  = lane & 15;

    const float*  Bv = mat ? B1 : B0;
    float*        Y  = mat ? H1 : H0;
    const __bf16* Wl = &lds_w[mat][0];

    v8f acc[8];
    #pragma unroll
    for (int ct = 0; ct < 8; ++ct) {
        float bval = Bv[ct * 16 + l16];   // col fixed per lane across all 8 acc rows
        #pragma unroll
        for (int r = 0; r < 8; ++r) acc[ct][r] = bval;
    }

    const __bf16* Xrow = Xbf + (size_t)(rowBase + l16) * DMID;   // A row = lane&15
    #pragma unroll
    for (int kc = 0; kc < 4; ++kc) {
        v16bf afrag = load_frag_bf16(Xrow, kc * 32, half);
        #pragma unroll
        for (int ct = 0; ct < 8; ++ct) {
            // B[k][n] = W[n][k] -> contiguous k-runs of row n of W (from LDS: ds_load_b128 x2)
            const __bf16* Wrow = Wl + (size_t)(ct * 16 + l16) * DMID;
            v16bf bfrag = load_frag_bf16(Wrow, kc * 32, half);
            acc[ct] = __builtin_amdgcn_wmma_f32_16x16x32_bf16(
                false, afrag, false, bfrag, (short)0, acc[ct], false, false);
        }
    }

    // C/D layout: VGPR r -> row half*8 + r, col = lane&15 (+16*ct)
    #pragma unroll
    for (int ct = 0; ct < 8; ++ct) {
        #pragma unroll
        for (int r = 0; r < 8; ++r) {
            int row = rowBase + half * 8 + r;
            Y[(size_t)row * DMID + ct * 16 + l16] = acc[ct][r];
        }
    }
}

// Weight f32 -> bf16 pre-conversion (weights are tiny; done once per launch).
__global__ __launch_bounds__(256) void f32_to_bf16_kernel(
    const float* __restrict__ src, __bf16* __restrict__ dst, int n)
{
    int i = blockIdx.x * 256 + threadIdx.x;
    if (i < n) dst[i] = (__bf16)src[i];
}

// Layer 0: di = 3, too small for WMMA K; scalar per (vertex, channel).
__global__ __launch_bounds__(256) void layer0_kernel(
    const float* __restrict__ verts,
    const float* __restrict__ w0, const float* __restrict__ b0,
    const float* __restrict__ w1, const float* __restrict__ b1,
    float* __restrict__ H0, float* __restrict__ H1)
{
    int v = blockIdx.x * 2 + (threadIdx.x >> 7);
    int c = threadIdx.x & 127;
    if (v >= N_VERTS) return;
    float x0 = verts[v * 3 + 0], x1 = verts[v * 3 + 1], x2 = verts[v * 3 + 2];
    H0[(size_t)v * DMID + c] = b0[c] + x0 * w0[c * 3 + 0] + x1 * w0[c * 3 + 1] + x2 * w0[c * 3 + 2];
    H1[(size_t)v * DMID + c] = b1[c] + x0 * w1[c * 3 + 0] + x1 * w1[c * 3 + 1] + x2 * w1[c * 3 + 2];
}

// Symmetric edge scatter: ACC[i] += H1[j]; ACC[j] += H1[i]. ACC pre-seeded with h0 (+bias),
// so after this pass ACC == x = h0 + agg. 128 consecutive threads per edge -> coalesced.
__global__ __launch_bounds__(256) void scatter_kernel(
    const float* __restrict__ H1, float* __restrict__ ACC, const int* __restrict__ edges)
{
    int e = blockIdx.x * 2 + (threadIdx.x >> 7);
    int c = threadIdx.x & 127;
    if (e >= N_EDGES) return;
    int i = edges[2 * e + 0];
    int j = edges[2 * e + 1];
    float vj = H1[(size_t)j * DMID + c];
    float vi = H1[(size_t)i * DMID + c];
    unsafeAtomicAdd(&ACC[(size_t)i * DMID + c], vj);
    unsafeAtomicAdd(&ACC[(size_t)j * DMID + c], vi);
}

// Per-channel sum / sum-of-squares partials, flushed with 2 atomics per block-channel.
__global__ __launch_bounds__(128) void bn_stats_kernel(
    const float* __restrict__ X, float* __restrict__ stats)
{
    int c = threadIdx.x;
    float s = 0.f, ss = 0.f;
    for (int r = blockIdx.x; r < N_VERTS; r += gridDim.x) {
        float v = X[(size_t)r * DMID + c];
        s += v; ss += v * v;
    }
    unsafeAtomicAdd(&stats[c], s);
    unsafeAtomicAdd(&stats[DMID + c], ss);
}

// BN + ReLU. Layers 0,1: write bf16 activations (next GEMM's A operand).
// Layer 2: write f32 (consumed by pooling).
__global__ __launch_bounds__(256) void bn_apply_kernel(
    const float* __restrict__ X, const float* __restrict__ stats,
    const float* __restrict__ gamma, const float* __restrict__ beta,
    float* __restrict__ outF, __bf16* __restrict__ outB)
{
    size_t idx = (size_t)blockIdx.x * 256 + threadIdx.x;
    int c = (int)(idx & 127);
    const float invN = 1.0f / (float)N_VERTS;
    float mu  = stats[c] * invN;
    float var = stats[DMID + c] * invN - mu * mu;
    float inv = rsqrtf(var + BN_EPS);
    float y = fmaxf(gamma[c] * (X[idx] - mu) * inv + beta[c], 0.f);
    if (outB) outB[idx] = (__bf16)y;
    else      outF[idx] = y;
}

// Segment sum exploiting sorted verts_idx: run-length accumulate, flush on mesh change.
#define POOL_R 128
__global__ __launch_bounds__(128) void pool_kernel(
    const float* __restrict__ X, const int* __restrict__ vid,
    float* __restrict__ pooled, float* __restrict__ cnt)
{
    int c  = threadIdx.x;
    int r0 = blockIdx.x * POOL_R;
    if (r0 >= N_VERTS) return;
    int r1 = r0 + POOL_R; if (r1 > N_VERTS) r1 = N_VERTS;
    int cur = vid[r0];
    float acc = 0.f, n = 0.f;
    for (int r = r0; r < r1; ++r) {
        int m = vid[r];
        if (m != cur) {
            unsafeAtomicAdd(&pooled[cur * DMID + c], acc);
            if (c == 0) unsafeAtomicAdd(&cnt[cur], n);
            acc = 0.f; n = 0.f; cur = m;
        }
        acc += X[(size_t)r * DMID + c];
        n   += 1.f;
    }
    unsafeAtomicAdd(&pooled[cur * DMID + c], acc);
    if (c == 0) unsafeAtomicAdd(&cnt[cur], n);
}

__global__ void pool_div_kernel(float* pooled, const float* cnt)
{
    int m = blockIdx.x, c = threadIdx.x;
    pooled[m * DMID + c] /= fmaxf(cnt[m], 1.0f);
}

__global__ __launch_bounds__(256) void fc1_kernel(
    const float* __restrict__ pooled, const float* __restrict__ w,
    const float* __restrict__ b, float* __restrict__ out)
{
    int m = blockIdx.x, o = threadIdx.x;
    const float* p  = pooled + m * DMID;
    const float* wr = w + o * DMID;
    float s = b[o];
    #pragma unroll 8
    for (int k = 0; k < DMID; ++k) s += p[k] * wr[k];
    out[m * NN1 + o] = fmaxf(s, 0.f);
}

__global__ void fc2_kernel(
    const float* __restrict__ hid, const float* __restrict__ w,
    const float* __restrict__ b, float* __restrict__ out, int hd)
{
    int m = blockIdx.x, c = threadIdx.x;
    if (c >= hd) return;
    const float* hr = hid + m * NN1;
    const float* wr = w + c * NN1;
    float s = b[c];
    for (int o = 0; o < NN1; ++o) s += hr[o] * wr[o];
    out[m * hd + c] = s;
}

extern "C" void kernel_launch(void* const* d_in, const int* in_sizes, int n_in,
                              void* d_out, int out_size, void* d_ws, size_t ws_size,
                              hipStream_t stream)
{
    const float* verts = (const float*)d_in[0];
    const int*   edges = (const int*)d_in[1];
    const int*   vid   = (const int*)d_in[2];
    const float* gw0[3] = {(const float*)d_in[3],  (const float*)d_in[4],  (const float*)d_in[5]};
    const float* gb0[3] = {(const float*)d_in[6],  (const float*)d_in[7],  (const float*)d_in[8]};
    const float* gw1[3] = {(const float*)d_in[9],  (const float*)d_in[10], (const float*)d_in[11]};
    const float* gb1[3] = {(const float*)d_in[12], (const float*)d_in[13], (const float*)d_in[14]};
    const float* gam[3] = {(const float*)d_in[15], (const float*)d_in[16], (const float*)d_in[17]};
    const float* bet[3] = {(const float*)d_in[18], (const float*)d_in[19], (const float*)d_in[20]};
    const float* f1w[4] = {(const float*)d_in[21], (const float*)d_in[22], (const float*)d_in[23], (const float*)d_in[24]};
    const float* f1b[4] = {(const float*)d_in[25], (const float*)d_in[26], (const float*)d_in[27], (const float*)d_in[28]};
    const float* f2w[4] = {(const float*)d_in[29], (const float*)d_in[30], (const float*)d_in[31], (const float*)d_in[32]};
    const float* f2b[4] = {(const float*)d_in[33], (const float*)d_in[34], (const float*)d_in[35], (const float*)d_in[36]};

    // Workspace layout (~307.2 MB + ~181 KB):
    char* ws = (char*)d_ws;
    const size_t XB = (size_t)N_VERTS * DMID * sizeof(float);
    float*  bufX   = (float*)(ws);            // f32 x (written only after layer 2)
    __bf16* Xbf    = (__bf16*)(ws);           // bf16 x, aliases bufX (disjoint live ranges)
    float*  bufH0  = (float*)(ws + XB);       // h0, becomes x = h0 + agg after scatter
    float*  bufH1  = (float*)(ws + 2 * XB);   // h1
    float*  stats  = (float*)(ws + 3 * XB);   // [128 sums | 128 sumsq]
    float*  pooled = stats + 2 * DMID;        // 32 x 128
    float*  cnt    = pooled + NUM_MESHES * DMID;  // 32
    float*  hid    = cnt + NUM_MESHES;        // 32 x 256 (reused per head)
    __bf16* wbf    = (__bf16*)(hid + NUM_MESHES * NN1);  // 4 x 128x128 bf16 weights

    // Pre-convert layer-1/2 weights to bf16 (4 x 16384 elements).
    for (int l = 1; l < 3; ++l) {
        f32_to_bf16_kernel<<<64, 256, 0, stream>>>(gw0[l], wbf + ((l - 1) * 2 + 0) * DMID * DMID, DMID * DMID);
        f32_to_bf16_kernel<<<64, 256, 0, stream>>>(gw1[l], wbf + ((l - 1) * 2 + 1) * DMID * DMID, DMID * DMID);
    }

    for (int l = 0; l < 3; ++l) {
        if (l == 0) {
            layer0_kernel<<<N_VERTS / 2, 256, 0, stream>>>(verts, gw0[0], gb0[0], gw1[0], gb1[0], bufH0, bufH1);
        } else {
            const __bf16* w0b = wbf + ((l - 1) * 2 + 0) * DMID * DMID;
            const __bf16* w1b = wbf + ((l - 1) * 2 + 1) * DMID * DMID;
            gemm_dual_wmma<<<N_VERTS / 64, 256, 0, stream>>>(Xbf, w0b, gb0[l], w1b, gb1[l], bufH0, bufH1);
        }

        scatter_kernel<<<N_EDGES / 2, 256, 0, stream>>>(bufH1, bufH0, edges);

        hipMemsetAsync(stats, 0, 2 * DMID * sizeof(float), stream);
        bn_stats_kernel<<<2048, 128, 0, stream>>>(bufH0, stats);
        if (l == 2)
            bn_apply_kernel<<<(N_VERTS * DMID) / 256, 256, 0, stream>>>(bufH0, stats, gam[l], bet[l], bufX, nullptr);
        else
            bn_apply_kernel<<<(N_VERTS * DMID) / 256, 256, 0, stream>>>(bufH0, stats, gam[l], bet[l], nullptr, Xbf);
    }

    hipMemsetAsync(pooled, 0, (NUM_MESHES * DMID + NUM_MESHES) * sizeof(float), stream);
    pool_kernel<<<(N_VERTS + POOL_R - 1) / POOL_R, 128, 0, stream>>>(bufX, vid, pooled, cnt);
    pool_div_kernel<<<NUM_MESHES, DMID, 0, stream>>>(pooled, cnt);

    float* out = (float*)d_out;
    const int hdims[4] = {3, 2, 4, 5};
    int off = 0;
    for (int h = 0; h < 4; ++h) {
        fc1_kernel<<<NUM_MESHES, NN1, 0, stream>>>(pooled, f1w[h], f1b[h], hid);
        fc2_kernel<<<NUM_MESHES, 32, 0, stream>>>(hid, f2w[h], f2b[h], out + off, hdims[h]);
        off += NUM_MESHES * hdims[h];
    }
}
